// Contrast_9397388444288
// MI455X (gfx1250) — matrix operand order repaired and checked by reading, not compiled, not run on hardware
//
#include <hip/hip_runtime.h>
#include <hip/hip_bf16.h>

#define N_ROWS 8192
#define DIM    256
#define ZSTR   512        // row stride in zs: [0..255]=hi bf16, [256..511]=lo bf16
#define HALF_B 4096

typedef __bf16 v16bf __attribute__((ext_vector_type(16)));
typedef __bf16 v8bf  __attribute__((ext_vector_type(8)));
typedef float  v8f   __attribute__((ext_vector_type(8)));

// ---------------------------------------------------------------------------
// Kernel 1: L2-normalize each row, split into bf16 hi/lo stored interleaved
// per row (stride ZSTR); zero pos/denom. One wave per row, 8 floats per lane.
// ---------------------------------------------------------------------------
__global__ __launch_bounds__(256) void nrm_split_kernel(
    const float* __restrict__ xi, const float* __restrict__ xj,
    __bf16* __restrict__ zs,
    float* __restrict__ pos, float* __restrict__ denom)
{
    const int wave = threadIdx.x >> 5;
    const int lane = threadIdx.x & 31;
    const int row  = blockIdx.x * 8 + wave;

    const float* src = (row < HALF_B) ? (xi + (size_t)row * DIM)
                                      : (xj + (size_t)(row - HALF_B) * DIM);
    float v[8];
    float ss = 0.0f;
#pragma unroll
    for (int i = 0; i < 8; ++i) { v[i] = src[lane * 8 + i]; ss += v[i] * v[i]; }
#pragma unroll
    for (int m = 16; m >= 1; m >>= 1) ss += __shfl_xor(ss, m, 32);

    const float scale = 1.0f / fmaxf(sqrtf(ss), 1e-12f);
    __bf16* dst = zs + (size_t)row * ZSTR + lane * 8;
#pragma unroll
    for (int i = 0; i < 8; ++i) {
        float z  = v[i] * scale;
        __bf16 h = (__bf16)z;                 // hi part
        __bf16 l = (__bf16)(z - (float)h);    // residual lo part
        dst[i]       = h;
        dst[256 + i] = l;
    }
    if (lane == 0) { denom[row] = 0.0f; pos[row] = 0.0f; }
}

// ---------------------------------------------------------------------------
// Kernel 2: fused sim = z z^T via bf16 hi/lo split WMMA, exp, row-sums, pos.
// Block = 8 waves; block tile 128x128; wave tile 16x128 (8 WMMA C tiles).
// B tiles processed in pairs with interleaved accumulator chains so dependent
// WMMAs are separated by an independent XDL op (kills the v_nop padding).
// ---------------------------------------------------------------------------
__global__ __launch_bounds__(256) void simexp_wmma_kernel(
    const __bf16* __restrict__ zs,
    float* __restrict__ pos, float* __restrict__ denom)
{
    const int wave = threadIdx.x >> 5;
    const int lane = threadIdx.x & 31;
    const int half = lane >> 4;       // which 16-lane half of the wave
    const int l15  = lane & 15;

    const int rowBase = blockIdx.y * 128 + wave * 16;
    const int colBase = blockIdx.x * 128;

    v8f acc[8] = {};                  // 8 x (16x16 f32) accumulators

    // A fragment: M = lane%16. this code's contiguous K order (B matches):
    //   lane<16 -> K 0..15, lane>=16 -> K 16..31
    const __bf16* aBase = zs + (size_t)(rowBase + l15) * ZSTR;
    // B fragment: column N = lane%16 = a row of z; lane half selects K 0..15 / 16..31.
    // One base pointer for the whole 128-col strip: tile t at +t*16*ZSTR (imm offset).
    const __bf16* bBase = zs + (size_t)(colBase + l15) * ZSTR;

    for (int kc = 0; kc < DIM; kc += 32) {
        const int ao = kc + 16 * half;    // A chunk offset
        const int bo = kc + 16 * half;    // B chunk offset
        union { v16bf v; v8bf h[2]; } aH, aL;
        aH.h[0] = *(const v8bf*)(aBase + ao);
        aH.h[1] = *(const v8bf*)(aBase + ao + 8);
        aL.h[0] = *(const v8bf*)(aBase + 256 + ao);
        aL.h[1] = *(const v8bf*)(aBase + 256 + ao + 8);

#pragma unroll
        for (int tp = 0; tp < 4; ++tp) {
            const int t0 = 2 * tp, t1 = 2 * tp + 1;
            const __bf16* b0 = bBase + (size_t)t0 * 16 * ZSTR + bo;
            const __bf16* b1 = bBase + (size_t)t1 * 16 * ZSTR + bo;
            union { v16bf v; v8bf h[2]; } b0H, b0L, b1H, b1L;
            b0H.h[0] = *(const v8bf*)(b0);
            b0H.h[1] = *(const v8bf*)(b0 + 8);
            b0L.h[0] = *(const v8bf*)(b0 + 256);
            b0L.h[1] = *(const v8bf*)(b0 + 256 + 8);
            b1H.h[0] = *(const v8bf*)(b1);
            b1H.h[1] = *(const v8bf*)(b1 + 8);
            b1L.h[0] = *(const v8bf*)(b1 + 256);
            b1L.h[1] = *(const v8bf*)(b1 + 256 + 8);

            // hi*hi + hi*lo + lo*hi (lo*lo ~2^-16, dropped), two chains interleaved
            acc[t0] = __builtin_amdgcn_wmma_f32_16x16x32_bf16(
                false, aL.v, false, b0H.v, (short)0, acc[t0], false, false);
            acc[t1] = __builtin_amdgcn_wmma_f32_16x16x32_bf16(
                false, aL.v, false, b1H.v, (short)0, acc[t1], false, false);
            acc[t0] = __builtin_amdgcn_wmma_f32_16x16x32_bf16(
                false, aH.v, false, b0L.v, (short)0, acc[t0], false, false);
            acc[t1] = __builtin_amdgcn_wmma_f32_16x16x32_bf16(
                false, aH.v, false, b1L.v, (short)0, acc[t1], false, false);
            acc[t0] = __builtin_amdgcn_wmma_f32_16x16x32_bf16(
                false, aH.v, false, b0H.v, (short)0, acc[t0], false, false);
            acc[t1] = __builtin_amdgcn_wmma_f32_16x16x32_bf16(
                false, aH.v, false, b1H.v, (short)0, acc[t1], false, false);
        }
    }

    // Epilogue. C/D layout: lanes 0-15 -> M=r, lanes 16-31 -> M=r+8; N = lane&15.
    float rowAcc[8];
#pragma unroll
    for (int r = 0; r < 8; ++r) rowAcc[r] = 0.0f;

#pragma unroll
    for (int t = 0; t < 8; ++t) {
#pragma unroll
        for (int r = 0; r < 8; ++r) {
            const float s    = acc[t][r];
            const int   gRow = rowBase + r + 8 * half;
            const int   gCol = colBase + t * 16 + l15;
            if (gRow != gCol) rowAcc[r] += expf(2.0f * s);   // sim / TEMP, TEMP = 0.5
            if (gCol == gRow + HALF_B || gCol + HALF_B == gRow) pos[gRow] = s;
        }
    }
#pragma unroll
    for (int r = 0; r < 8; ++r) {
        float v = rowAcc[r];
#pragma unroll
        for (int m = 8; m >= 1; m >>= 1) v += __shfl_xor(v, m, 16);
        if (l15 == 0)
            unsafeAtomicAdd(&denom[rowBase + r + 8 * half], v);  // global_atomic_add_f32
    }
}

// ---------------------------------------------------------------------------
// Kernel 3: loss = sum(-pos/T + log(denom)) / (2B)
// ---------------------------------------------------------------------------
__global__ __launch_bounds__(256) void finalize_kernel(
    const float* __restrict__ pos, const float* __restrict__ denom,
    float* __restrict__ out)
{
    __shared__ float red[256];
    float p = 0.0f;
    for (int i = threadIdx.x; i < N_ROWS; i += 256)
        p += -2.0f * pos[i] + logf(denom[i]);
    red[threadIdx.x] = p;
    __syncthreads();
    for (int s = 128; s > 0; s >>= 1) {
        if ((int)threadIdx.x < s) red[threadIdx.x] += red[threadIdx.x + s];
        __syncthreads();
    }
    if (threadIdx.x == 0) out[0] = red[0] / (float)N_ROWS;
}

// ---------------------------------------------------------------------------
extern "C" void kernel_launch(void* const* d_in, const int* in_sizes, int n_in,
                              void* d_out, int out_size, void* d_ws, size_t ws_size,
                              hipStream_t stream) {
    const float* xi = (const float*)d_in[0];
    const float* xj = (const float*)d_in[1];
    float* out = (float*)d_out;

    // Workspace carve-out (~8.06 MB total)
    __bf16* zs   = (__bf16*)d_ws;                        // 8192 * 512 bf16 = 8 MB
    float*  pos   = (float*)(zs + (size_t)N_ROWS * ZSTR); // 32 KB
    float*  denom = pos + N_ROWS;                         // 32 KB

    nrm_split_kernel<<<N_ROWS / 8, 256, 0, stream>>>(xi, xj, zs, pos, denom);

    dim3 grid(N_ROWS / 128, N_ROWS / 128);   // 64 x 64 blocks of 128x128
    simexp_wmma_kernel<<<grid, 256, 0, stream>>>(zs, pos, denom);

    finalize_kernel<<<1, 256, 0, stream>>>(pos, denom, out);
}
